// cnn_8950711845742
// MI455X (gfx1250) — matrix operand order repaired and checked by reading, not compiled, not run on hardware
//
#include <hip/hip_runtime.h>

typedef __attribute__((ext_vector_type(16))) _Float16 v16h;
typedef __attribute__((ext_vector_type(8)))  float    v8f;

#define NSTEP  300
#define THRESH 1.0f
#define DECM   0.95122942450071400910f   // exp(-1/20)
#define DECS   0.81873075307798185867f   // exp(-1/5)

union AB16 { v16h v; _Float16 h[16]; unsigned u[8]; uint4 q[2]; };

// One workgroup per batch sample; entire 300-step recurrence stays on-chip.
// State & weights LDS-resident; convs are implicit GEMMs on v_wmma_f32_16x16x32_f16.
// No WMMA sits under any divergent/EXEC-masked path (ISA: EXEC must be all ones).
__global__ __launch_bounds__(256, 1)
void noso_step_kernel(const float* __restrict__ xin,
                      const float* __restrict__ w1,
                      const float* __restrict__ w2,
                      const float* __restrict__ wfc,
                      float* __restrict__ out,
                      float* __restrict__ ws,
                      int B)
{
    // ---- LDS (~263 KB total, fits 320 KB/WGP on CDNA5) ----
    __shared__ __align__(16) _Float16 s_w1h[16 * 64];    //  2 KB  conv1 W, M pad 12->16, K pad 50->64
    __shared__ __align__(16) _Float16 s_w2h[64 * 320];   // 40 KB  conv2 W, K pad 300->320
    __shared__ __align__(16) _Float16 s_xh[2080];        //  4 KB  input spikes (2,32,32) in f16 (+pad)
    // layer1 traces, neuron-major [n][co], stride 12 floats (48B: bank-conflict-free); n padded to 800
    __shared__ float    s_cm1[800 * 12], s_cs1[800 * 12], s_sv1[800 * 12];   // 112.5 KB
    // layer2 traces, neuron-major [n][co], stride 68 floats (272B: avoids 256B bank period)
    __shared__ float    s_cm2[100 * 68], s_cs2[100 * 68], s_sv2[100 * 68];   //  80 KB
    __shared__ unsigned s_p1[2352];                      //  9 KB  pooled sp1 (12,14,14); f16 1.0 in low half
    __shared__ unsigned s_xf[1600];                      //  6 KB  pooled sp2 flat (64,5,5); f32 1.0 bits
    __shared__ int      s_kb1[64];                       // im2col base offsets conv1
    __shared__ int      s_kb2[320];                      // im2col base offsets conv2
    __shared__ float    s_cmf[10], s_csf[10], s_svf[10], s_ot[10], s_ou[10];
    __shared__ float    s_spf;
    __shared__ float    s_red[256];

    const int tid = threadIdx.x;
    const int b   = blockIdx.x;
    const int wv  = tid >> 5;   // wave id (8 waves)
    const int ln  = tid & 31;   // lane in wave32
    const int hw  = ln >> 4;    // half-wave
    const int lm  = ln & 15;    // lane mod 16

    // ---- one-time init ----
    for (int i = tid; i < 16 * 64; i += 256) {
        int co = i >> 6, k = i & 63;
        s_w1h[i] = (co < 12 && k < 50) ? (_Float16)w1[co * 50 + k] : (_Float16)0.0f;
    }
    for (int i = tid; i < 64 * 320; i += 256) {
        int co = i / 320, k = i - co * 320;
        s_w2h[i] = (k < 300) ? (_Float16)w2[co * 300 + k] : (_Float16)0.0f;
    }
    for (int i = tid; i < 800 * 12; i += 256) { s_cm1[i] = 0.f; s_cs1[i] = 0.f; s_sv1[i] = 1.f; }
    for (int i = tid; i < 100 * 68; i += 256) { s_cm2[i] = 0.f; s_cs2[i] = 0.f; s_sv2[i] = 1.f; }
    if (tid < 32) s_xh[2048 + tid] = (_Float16)0.0f;     // zero the gather pad (fake tiles)
    if (tid < 64)
        s_kb1[tid] = (tid < 50) ? ((tid / 25) * 32 + ((tid % 25) / 5)) * 32 + (tid % 5) : 0;
    for (int i = tid; i < 320; i += 256)
        s_kb2[i] = (i < 300) ? ((i / 25) * 14 + ((i % 25) / 5)) * 14 + (i % 5) : 0;
    if (tid < 10) { s_cmf[tid] = 0.f; s_csf[tid] = 0.f; s_svf[tid] = 1.f;
                    s_ot[tid] = (float)NSTEP; s_ou[tid] = 0.f; }
    if (tid == 0) s_spf = 0.f;

    float c_in = 0.f, c_s1 = 0.f, c_s2 = 0.f;
    const float* xb = xin + (size_t)b * (2u * 34u * 34u * 300u);

    // fused NOSO trace update + spike + pool(OR); `valid` gates side effects for N-padding
    auto upd1 = [&](float& cm, float& cs, float& sv, float a, int co, int poff, bool valid) {
        cm = DECM * cm + a;
        cs = DECS * cs + a;
        float u  = sv * (cm - cs);
        float sp = (u > THRESH) ? 1.f : 0.f;
        sv *= (1.f - sp);
        if (valid) c_s1 += sp;
        if (valid && sp > 0.f) atomicOr(&s_p1[co * 196 + poff], 0x00003C00u); // f16 1.0 low half
    };
    auto upd2 = [&](float& cm, float& cs, float& sv, float a, int co, int xoff2) {
        cm = DECM * cm + a;
        cs = DECS * cs + a;
        float u  = sv * (cm - cs);
        float sp = (u > THRESH) ? 1.f : 0.f;
        sv *= (1.f - sp);
        c_s2 += sp;
        if (sp > 0.f) atomicOr(&s_xf[co * 25 + xoff2], 0x3F800000u);         // f32 1.0 bits
    };

    for (int t = 0; t < NSTEP; ++t) {
        __syncthreads();   // previous step's consumers done; init done on t==0
        for (int i = tid; i < 2352; i += 256) s_p1[i] = 0u;
        for (int i = tid; i < 1600; i += 256) s_xf[i] = 0u;
        // stage cropped input slice [1:33,1:33] as f16; t is fastest-varying in memory, so a
        // 128B line holds 32 consecutive steps -> L2-resident reuse; prefetch next line.
        for (int i = tid; i < 2048; i += 256) {
            int c = i >> 10, r = i & 1023, y = r >> 5, x = r & 31;
            size_t idx = (((size_t)c * 34 + (y + 1)) * 34 + (x + 1)) * 300 + t;
            float v = xb[idx];
            if ((t & 31) == 0) __builtin_prefetch(&xb[idx + 32], 0, 0);
            s_xh[i] = (_Float16)v; c_in += v;
        }
        __syncthreads();

        // ---- layer 1: conv1 implicit GEMM, M=16(12) x N=800(784) x K=64(50) ----
        // 25 unconditional N-tile pairs (N padded to 800); A fragment shared per pair.
        for (int np = wv; np < 25; np += 8) {            // wave-uniform bounds
            int n0 = np * 32 + lm;                       // always < 784 (valid)
            int n1 = n0 + 16;                            // fake only when np==24 (784..799)
            int oy0 = n0 / 28, ox0 = n0 % 28, xoff0 = oy0 * 32 + ox0;
            int oy1 = n1 / 28, ox1 = n1 % 28, xoff1 = oy1 * 32 + ox1;
            v8f acc0 = {}, acc1 = {};
            #pragma unroll
            for (int kit = 0; kit < 2; ++kit) {
                int kb = kit * 32;
                AB16 A, B0, B1;
                const _Float16* wrow = &s_w1h[lm << 6];
                A.q[0] = *(const uint4*)&wrow[kb + (hw << 4)];        // 8 contiguous f16
                A.q[1] = *(const uint4*)&wrow[kb + (hw << 4) + 8];
                #pragma unroll
                for (int v2 = 0; v2 < 8; ++v2) {
                    int kk = kb + (hw << 4) + (v2 << 1);
                    int o0 = s_kb1[kk], o1 = s_kb1[kk + 1];
                    B0.h[2 * v2]     = s_xh[o0 + xoff0];              // direct f16 DS loads
                    B0.h[2 * v2 + 1] = s_xh[o1 + xoff0];
                    B1.h[2 * v2]     = s_xh[o0 + xoff1];
                    B1.h[2 * v2 + 1] = s_xh[o1 + xoff1];
                }
                acc0 = __builtin_amdgcn_wmma_f32_16x16x32_f16(
                           false, A.v, false, B0.v, (short)0, acc0, false, false);
                acc1 = __builtin_amdgcn_wmma_f32_16x16x32_f16(
                           false, A.v, false, B1.v, (short)0, acc1, false, false);
            }
            // NOSO epilogue: float4 state chunks; co = 8*hw + r
            {
                int poff = (oy0 >> 1) * 14 + (ox0 >> 1);
                int base = n0 * 12 + (hw << 3);
                float4 cm4 = *(float4*)&s_cm1[base];
                float4 cs4 = *(float4*)&s_cs1[base];
                float4 sv4 = *(float4*)&s_sv1[base];
                int c0 = hw << 3;
                upd1(cm4.x, cs4.x, sv4.x, acc0[0], c0 + 0, poff, true);
                upd1(cm4.y, cs4.y, sv4.y, acc0[1], c0 + 1, poff, true);
                upd1(cm4.z, cs4.z, sv4.z, acc0[2], c0 + 2, poff, true);
                upd1(cm4.w, cs4.w, sv4.w, acc0[3], c0 + 3, poff, true);
                *(float4*)&s_cm1[base] = cm4; *(float4*)&s_cs1[base] = cs4; *(float4*)&s_sv1[base] = sv4;
                if (hw == 0) {                           // co 4..7 (co 12..15 don't exist)
                    int b2 = n0 * 12 + 4;
                    float4 dm4 = *(float4*)&s_cm1[b2];
                    float4 ds4 = *(float4*)&s_cs1[b2];
                    float4 dv4 = *(float4*)&s_sv1[b2];
                    upd1(dm4.x, ds4.x, dv4.x, acc0[4], 4, poff, true);
                    upd1(dm4.y, ds4.y, dv4.y, acc0[5], 5, poff, true);
                    upd1(dm4.z, ds4.z, dv4.z, acc0[6], 6, poff, true);
                    upd1(dm4.w, ds4.w, dv4.w, acc0[7], 7, poff, true);
                    *(float4*)&s_cm1[b2] = dm4; *(float4*)&s_cs1[b2] = ds4; *(float4*)&s_sv1[b2] = dv4;
                }
            }
            {
                bool val1 = (n1 < 784);
                int poff = (oy1 >> 1) * 14 + (ox1 >> 1);
                int base = n1 * 12 + (hw << 3);
                float4 cm4 = *(float4*)&s_cm1[base];
                float4 cs4 = *(float4*)&s_cs1[base];
                float4 sv4 = *(float4*)&s_sv1[base];
                int c0 = hw << 3;
                upd1(cm4.x, cs4.x, sv4.x, acc1[0], c0 + 0, poff, val1);
                upd1(cm4.y, cs4.y, sv4.y, acc1[1], c0 + 1, poff, val1);
                upd1(cm4.z, cs4.z, sv4.z, acc1[2], c0 + 2, poff, val1);
                upd1(cm4.w, cs4.w, sv4.w, acc1[3], c0 + 3, poff, val1);
                *(float4*)&s_cm1[base] = cm4; *(float4*)&s_cs1[base] = cs4; *(float4*)&s_sv1[base] = sv4;
                if (hw == 0) {
                    int b2 = n1 * 12 + 4;
                    float4 dm4 = *(float4*)&s_cm1[b2];
                    float4 ds4 = *(float4*)&s_cs1[b2];
                    float4 dv4 = *(float4*)&s_sv1[b2];
                    upd1(dm4.x, ds4.x, dv4.x, acc1[4], 4, poff, val1);
                    upd1(dm4.y, ds4.y, dv4.y, acc1[5], 5, poff, val1);
                    upd1(dm4.z, ds4.z, dv4.z, acc1[6], 6, poff, val1);
                    upd1(dm4.w, ds4.w, dv4.w, acc1[7], 7, poff, val1);
                    *(float4*)&s_cm1[b2] = dm4; *(float4*)&s_cs1[b2] = ds4; *(float4*)&s_sv1[b2] = dv4;
                }
            }
        }
        __syncthreads();

        // ---- layer 2: conv2 implicit GEMM, M=64 x N=128(100) x K=320(300) ----
        // One N-tile per wave (tile 7 is pure padding -> no wave guard, no EXEC-masked WMMA);
        // B fragment shared across all 4 M-tiles (4 independent accumulators).
        {
            int n  = wv * 16 + lm;
            int ns = (n < 100) ? n : 99;                 // branchless clamp keeps EXEC full
            int oy = ns / 10, ox = ns - oy * 10;
            int poff = oy * 14 + ox;
            const _Float16* p1h = (const _Float16*)s_p1; // f16 value lives in low half of each word
            v8f z = {};
            v8f acc[4] = { z, z, z, z };
            for (int kit = 0; kit < 10; ++kit) {
                int kb = kit * 32;
                AB16 Bm;
                #pragma unroll
                for (int v2 = 0; v2 < 8; ++v2) {
                    int kk = kb + (hw << 4) + (v2 << 1);
                    Bm.h[2 * v2]     = p1h[(s_kb2[kk]     + poff) << 1];
                    Bm.h[2 * v2 + 1] = p1h[(s_kb2[kk + 1] + poff) << 1];
                }
                #pragma unroll
                for (int mt = 0; mt < 4; ++mt) {
                    AB16 A;
                    const _Float16* wrow = &s_w2h[(mt * 16 + lm) * 320];
                    A.q[0] = *(const uint4*)&wrow[kb + (hw << 4)];
                    A.q[1] = *(const uint4*)&wrow[kb + (hw << 4) + 8];
                    acc[mt] = __builtin_amdgcn_wmma_f32_16x16x32_f16(
                                  false, A.v, false, Bm.v, (short)0, acc[mt], false, false);
                }
            }
            if (n < 100) {
                int xoff2 = (oy >> 1) * 5 + (ox >> 1);
                #pragma unroll
                for (int mt = 0; mt < 4; ++mt) {
                    int c0   = mt * 16 + (hw << 3);      // co = c0 + r, all valid (co < 64)
                    int base = n * 68 + c0;
                    float4 cm4 = *(float4*)&s_cm2[base];
                    float4 cs4 = *(float4*)&s_cs2[base];
                    float4 sv4 = *(float4*)&s_sv2[base];
                    upd2(cm4.x, cs4.x, sv4.x, acc[mt][0], c0 + 0, xoff2);
                    upd2(cm4.y, cs4.y, sv4.y, acc[mt][1], c0 + 1, xoff2);
                    upd2(cm4.z, cs4.z, sv4.z, acc[mt][2], c0 + 2, xoff2);
                    upd2(cm4.w, cs4.w, sv4.w, acc[mt][3], c0 + 3, xoff2);
                    *(float4*)&s_cm2[base] = cm4; *(float4*)&s_cs2[base] = cs4; *(float4*)&s_sv2[base] = sv4;
                    int b2 = base + 4;
                    float4 dm4 = *(float4*)&s_cm2[b2];
                    float4 ds4 = *(float4*)&s_cs2[b2];
                    float4 dv4 = *(float4*)&s_sv2[b2];
                    upd2(dm4.x, ds4.x, dv4.x, acc[mt][4], c0 + 4, xoff2);
                    upd2(dm4.y, ds4.y, dv4.y, acc[mt][5], c0 + 5, xoff2);
                    upd2(dm4.z, ds4.z, dv4.z, acc[mt][6], c0 + 6, xoff2);
                    upd2(dm4.w, ds4.w, dv4.w, acc[mt][7], c0 + 7, xoff2);
                    *(float4*)&s_cm2[b2] = dm4; *(float4*)&s_cs2[b2] = ds4; *(float4*)&s_sv2[b2] = dv4;
                }
            }
        }
        __syncthreads();

        // ---- FC layer: one wave per output, wave32 shuffle reduction ----
        for (int pass = 0; pass < 2; ++pass) {
            int o = (pass == 0) ? wv : 8 + wv;
            if (o < 10) {
                const float* wrow = wfc + o * 1600;
                float p = 0.f;
                for (int k = ln; k < 1600; k += 32)
                    p += __uint_as_float(s_xf[k]) * wrow[k];
                #pragma unroll
                for (int m = 16; m >= 1; m >>= 1) p += __shfl_xor(p, m, 32);
                if (ln == 0) {
                    float cm = DECM * s_cmf[o] + p;
                    float cs = DECS * s_csf[o] + p;
                    float sv = s_svf[o];
                    float u  = sv * (cm - cs);
                    float sp = (u > THRESH) ? 1.f : 0.f;
                    s_cmf[o] = cm; s_csf[o] = cs; s_svf[o] = sv * (1.f - sp);
                    s_ot[o] += sp * ((float)t - (float)NSTEP);
                    s_ou[o] += sp * u;
                    if (sp > 0.f) atomicAdd(&s_spf, 1.f);
                }
            }
        }
    }
    __syncthreads();

    // ---- block reductions of spike counters ----
    s_red[tid] = c_in; __syncthreads();
    for (int s = 128; s > 0; s >>= 1) { if (tid < s) s_red[tid] += s_red[tid + s]; __syncthreads(); }
    float tot_in = s_red[0]; __syncthreads();
    s_red[tid] = c_s1; __syncthreads();
    for (int s = 128; s > 0; s >>= 1) { if (tid < s) s_red[tid] += s_red[tid + s]; __syncthreads(); }
    float tot_s1 = s_red[0]; __syncthreads();
    s_red[tid] = c_s2; __syncthreads();
    for (int s = 128; s > 0; s >>= 1) { if (tid < s) s_red[tid] += s_red[tid + s]; __syncthreads(); }
    float tot_s2 = s_red[0];

    if (tid == 0) {
        ws[b * 4 + 0] = tot_in; ws[b * 4 + 1] = tot_s1;
        ws[b * 4 + 2] = tot_s2; ws[b * 4 + 3] = s_spf;
    }
    if (tid < 10) {
        out[b * 10 + tid]          = s_ot[tid];   // out_t
        out[B * 10 + b * 10 + tid] = s_ou[tid];   // out_u
    }
}

// Deterministic cross-sample combine for sum_sp (avoids float atomic ordering).
__global__ void noso_final_kernel(const float* __restrict__ ws, float* __restrict__ out, int B)
{
    int j = threadIdx.x;
    if (j < 4) {
        float s = 0.f;
        for (int b = 0; b < B; ++b) s += ws[b * 4 + j];
        out[2 * B * 10 + j] = s;
    }
}

extern "C" void kernel_launch(void* const* d_in, const int* in_sizes, int n_in,
                              void* d_out, int out_size, void* d_ws, size_t ws_size,
                              hipStream_t stream)
{
    (void)n_in; (void)out_size; (void)ws_size;
    const float* xin = (const float*)d_in[0];   // input spikes (B,2,34,34,300)
    const float* w1  = (const float*)d_in[2];   // (12,2,5,5)
    const float* w2  = (const float*)d_in[3];   // (64,12,5,5)
    const float* wfc = (const float*)d_in[4];   // (10,1600)
    int B = in_sizes[0] / (2 * 34 * 34 * 300);

    noso_step_kernel<<<B, 256, 0, stream>>>(xin, w1, w2, wfc,
                                            (float*)d_out, (float*)d_ws, B);
    noso_final_kernel<<<1, 32, 0, stream>>>((const float*)d_ws, (float*)d_out, B);
}
